// MixedPrecisionAttention_20590073217574
// MI455X (gfx1250) — matrix operand order repaired and checked by reading, not compiled, not run on hardware
//
#include <hip/hip_runtime.h>
#include <math.h>

typedef __attribute__((ext_vector_type(16))) _Float16 v16h;
typedef __attribute__((ext_vector_type(8)))  _Float16 v8h;
typedef __attribute__((ext_vector_type(8)))  float    v8f;
typedef __attribute__((ext_vector_type(4)))  float    v4f;

namespace {

constexpr int Bc = 4, Sc = 2048, Hc = 8, Dc = 64, Cc = Hc * Dc;   // C = 512
constexpr int KBLK  = 32;            // keys per block iteration
constexpr int NBLK  = Sc / KBLK;     // 64 key blocks
constexpr int WAVES = 8;             // waves per workgroup (wave32)
constexpr float SCALING = 0.125f;    // 64^-0.5

union Frag16 { v16h v; v8h h[2]; };

// load 8 consecutive f32 and convert to packed f16 (two b128 loads + cvt_pk)
__device__ __forceinline__ v8h cvt8(const float* p) {
  v4f a = *(const v4f*)p;
  v4f b = *(const v4f*)(p + 4);
  v8h r;
  r[0] = (_Float16)a[0]; r[1] = (_Float16)a[1];
  r[2] = (_Float16)a[2]; r[3] = (_Float16)a[3];
  r[4] = (_Float16)b[0]; r[5] = (_Float16)b[1];
  r[6] = (_Float16)b[2]; r[7] = (_Float16)b[3];
  return r;
}

// reductions across the 16-lane halves of the wave (xor masks < 16 never
// cross the half boundary; matches the 16x16 C/D tile row layout)
__device__ __forceinline__ float redmax16(float v) {
  v = fmaxf(v, __shfl_xor(v, 1, 32));
  v = fmaxf(v, __shfl_xor(v, 2, 32));
  v = fmaxf(v, __shfl_xor(v, 4, 32));
  v = fmaxf(v, __shfl_xor(v, 8, 32));
  return v;
}
__device__ __forceinline__ float redsum16(float v) {
  v += __shfl_xor(v, 1, 32);
  v += __shfl_xor(v, 2, 32);
  v += __shfl_xor(v, 4, 32);
  v += __shfl_xor(v, 8, 32);
  return v;
}

} // namespace

__global__ __launch_bounds__(256)
void mpattn_kernel(const float* __restrict__ Q, const float* __restrict__ Kt,
                   const float* __restrict__ V,
                   const float* __restrict__ p_sqk, const float* __restrict__ p_zqk,
                   const float* __restrict__ p_sat, const float* __restrict__ p_zat,
                   const float* __restrict__ p_sm,  float* __restrict__ Out)
{
  // double-buffered K tile: [buf][key][dd], row stride 72 f16 = 144 B (16B aligned)
  __shared__ __align__(16) _Float16 Ks[2][KBLK][72];
  // double-buffered V tile (transposed): [buf][dd][key], row stride 40 f16 = 80 B
  __shared__ __align__(16) _Float16 Vs[2][Dc][40];
  // per-wave P staging: 16 rows x 32 keys, row stride 64 B
  __shared__ __align__(16) _Float16 Ps[WAVES][16][32];

  const int tid  = threadIdx.x;
  const int wave = tid >> 5;
  const int lane = tid & 31;
  const int half = lane >> 4;
  const int l16  = lane & 15;

  const int wg   = blockIdx.x;
  const int qblk = wg & 15;          // Sc / 128 = 16 q-blocks
  const int h    = (wg >> 4) & 7;
  const int b    = wg >> 7;
  const int q0   = qblk * 128 + wave * 16;

  // hoist quantization constants: no divisions inside the key loop
  const float sqk = p_sqk[0], zqk = p_zqk[0];
  const float sat = p_sat[0], zat = p_zat[0];
  const float sms = p_sm[0];
  const float qa  = SCALING / sqk;   // acc -> quant grid (one divide, hoisted)
  const float qc  = sqk * sms;       // dequant * softmax_scale, folded
  const float pa  = 1.0f / sat;      // prob -> quant grid

  // ---- Q fragments (A-layout f16): two 16x32 tiles covering d=64 ----
  Frag16 qf[2];
  {
    const int row = q0 + l16;
    const float* qb = Q + ((size_t)b * Sc + row) * Cc + (size_t)h * Dc;
    #pragma unroll
    for (int t = 0; t < 2; ++t) {
      const int c0 = t * 32;
      // lanes 0-15: K = {0..15}; lanes 16-31: K = {16..31}
      qf[t].h[0] = cvt8(qb + c0 + half * 16);
      qf[t].h[1] = cvt8(qb + c0 + half * 16 + 8);
    }
  }

  const size_t kbase = (size_t)b * Cc * Sc;
  const size_t vbase = (size_t)b * Sc * Cc;

  // per-thread cooperative-fill coordinates
  const int kks = tid & 31, kdd0 = tid >> 5;   // K tile: key = kks, dd = kdd0+8j
  const int vdd = tid & 63, vks0 = tid >> 6;   // V tile: dd = vdd, key = vks0+4j

  // lane-local running max / sum (two-level online softmax: no shuffles
  // inside the key loop, one cross-lane combine at the end)
  float m[8], l[8];
  #pragma unroll
  for (int i = 0; i < 8; ++i) { m[i] = -INFINITY; l[i] = 0.0f; }

  // ================= pass 1: lane-local online max / sum =================
  {
    // prologue: fill buffer 0
    float rk[8];
    #pragma unroll
    for (int j = 0; j < 8; ++j)
      rk[j] = Kt[kbase + ((size_t)(kdd0 + j * 8) * Hc + h) * Sc + kks];
    #pragma unroll
    for (int j = 0; j < 8; ++j)
      Ks[0][kks][kdd0 + j * 8] = (_Float16)rk[j];
  }
  __syncthreads();

  for (int blk = 0; blk < NBLK; ++blk) {
    const int s0  = blk * KBLK;
    const int cur = blk & 1, nxt = cur ^ 1;
    const bool more = (blk + 1 < NBLK);

    // (1) issue next tile's global loads into registers (latency hidden by WMMA)
    float rk[8];
    if (more) {
      #pragma unroll
      for (int j = 0; j < 8; ++j)
        rk[j] = Kt[kbase + ((size_t)(kdd0 + j * 8) * Hc + h) * Sc + (s0 + KBLK + kks)];
    }

    // (2) compute on current buffer
    float sc[2][8];
    #pragma unroll
    for (int nt = 0; nt < 2; ++nt) {
      const int key = nt * 16 + l16;      // B-layout: lane = column (key)
      v8f acc = {};
      #pragma unroll
      for (int c = 0; c < 2; ++c) {
        Frag16 bf;
        const int ddb = c * 32 + half * 16;  // halves split reduction dim
        bf.h[0] = *(const v8h*)&Ks[cur][key][ddb];
        bf.h[1] = *(const v8h*)&Ks[cur][key][ddb + 8];
        acc = __builtin_amdgcn_wmma_f32_16x16x32_f16(false, qf[c].v, false, bf.v,
                                                     (short)0, acc, false, false);
      }
      #pragma unroll
      for (int i = 0; i < 8; ++i) {
        float q = rintf(acc[i] * qa) + zqk;      // 4-bit fake-quant, no div
        q = fminf(fmaxf(q, 0.0f), 15.0f);
        sc[nt][i] = (q - zqk) * qc;
      }
    }
    #pragma unroll
    for (int i = 0; i < 8; ++i) {
      const float tm = fmaxf(sc[0][i], sc[1][i]);
      const float mn = fmaxf(m[i], tm);
      l[i] = l[i] * __expf(m[i] - mn)
           + __expf(sc[0][i] - mn) + __expf(sc[1][i] - mn);
      m[i] = mn;
    }

    // (3) cvt + store staged registers into the other buffer
    if (more) {
      #pragma unroll
      for (int j = 0; j < 8; ++j)
        Ks[nxt][kks][kdd0 + j * 8] = (_Float16)rk[j];
    }
    // (4) single barrier per iteration
    __syncthreads();
  }

  // cross-lane combine: row max + rescaled row sum (8 shuffles per row, once)
  float rl[8];
  #pragma unroll
  for (int i = 0; i < 8; ++i) {
    const float mr = redmax16(m[i]);
    const float lr = redsum16(l[i] * __expf(m[i] - mr));
    m[i]  = mr;
    rl[i] = 1.0f / lr;
  }

  v8f o[4] = {};

  // ========== pass 2: p = fq3(softmax) ; O += P x V (WMMA) ==========
  {
    // prologue: fill buffer 0 with K and V tiles of block 0
    float rk[8], rv[8];
    #pragma unroll
    for (int j = 0; j < 8; ++j)
      rk[j] = Kt[kbase + ((size_t)(kdd0 + j * 8) * Hc + h) * Sc + kks];
    #pragma unroll
    for (int j = 0; j < 8; ++j)
      rv[j] = V[vbase + (size_t)(vks0 + j * 4) * Cc + (size_t)h * Dc + vdd];
    #pragma unroll
    for (int j = 0; j < 8; ++j)
      Ks[0][kks][kdd0 + j * 8] = (_Float16)rk[j];
    #pragma unroll
    for (int j = 0; j < 8; ++j)
      Vs[0][vdd][vks0 + j * 4] = (_Float16)rv[j];
  }
  __syncthreads();

  for (int blk = 0; blk < NBLK; ++blk) {
    const int s0  = blk * KBLK;
    const int cur = blk & 1, nxt = cur ^ 1;
    const bool more = (blk + 1 < NBLK);

    // (1) issue next K/V tile loads
    float rk[8], rv[8];
    if (more) {
      #pragma unroll
      for (int j = 0; j < 8; ++j)
        rk[j] = Kt[kbase + ((size_t)(kdd0 + j * 8) * Hc + h) * Sc + (s0 + KBLK + kks)];
      #pragma unroll
      for (int j = 0; j < 8; ++j)
        rv[j] = V[vbase + (size_t)(s0 + KBLK + vks0 + j * 4) * Cc + (size_t)h * Dc + vdd];
    }

    // (2) compute: scores -> fq4 -> softmax -> fq3 -> P x V
    #pragma unroll
    for (int nt = 0; nt < 2; ++nt) {
      const int key = nt * 16 + l16;
      v8f acc = {};
      #pragma unroll
      for (int c = 0; c < 2; ++c) {
        Frag16 bf;
        const int ddb = c * 32 + half * 16;
        bf.h[0] = *(const v8h*)&Ks[cur][key][ddb];
        bf.h[1] = *(const v8h*)&Ks[cur][key][ddb + 8];
        acc = __builtin_amdgcn_wmma_f32_16x16x32_f16(false, qf[c].v, false, bf.v,
                                                     (short)0, acc, false, false);
      }
      #pragma unroll
      for (int i = 0; i < 8; ++i) {
        float q = rintf(acc[i] * qa) + zqk;          // 4-bit fq on scores
        q = fminf(fmaxf(q, 0.0f), 15.0f);
        const float s = (q - zqk) * qc;
        const float p = __expf(s - m[i]) * rl[i];    // final normalized prob
        float pq = rintf(p * pa) + zat;              // 3-bit fq on probs
        pq = fminf(fmaxf(pq, 0.0f), 7.0f);
        pq = (pq - zat) * sat;
        // D-layout -> LDS: VGPR i is row (i + half*8), N = l16
        Ps[wave][i + half * 8][nt * 16 + l16] = (_Float16)pq;
      }
    }

    // Re-read P in A-layout (same wave wrote it; DS ops stay in order)
    Frag16 pf;
    pf.h[0] = *(const v8h*)&Ps[wave][l16][half * 16];
    pf.h[1] = *(const v8h*)&Ps[wave][l16][half * 16 + 8];

    #pragma unroll
    for (int nt = 0; nt < 4; ++nt) {
      Frag16 vf;
      const int dd = nt * 16 + l16;       // B-layout: lane = output dim column
      vf.h[0] = *(const v8h*)&Vs[cur][dd][half * 16];
      vf.h[1] = *(const v8h*)&Vs[cur][dd][half * 16 + 8];
      o[nt] = __builtin_amdgcn_wmma_f32_16x16x32_f16(false, pf.v, false, vf.v,
                                                     (short)0, o[nt], false, false);
    }

    // (3) store staged K/V into the other buffer
    if (more) {
      #pragma unroll
      for (int j = 0; j < 8; ++j)
        Ks[nxt][kks][kdd0 + j * 8] = (_Float16)rk[j];
      #pragma unroll
      for (int j = 0; j < 8; ++j)
        Vs[nxt][vdd][vks0 + j * 4] = (_Float16)rv[j];
    }
    // (4) single barrier per iteration
    __syncthreads();
  }

  // ---- store O: out[b, q0+M, h*64 + nt*16 + N] from D-layout ----
  #pragma unroll
  for (int nt = 0; nt < 4; ++nt) {
    #pragma unroll
    for (int i = 0; i < 8; ++i) {
      const int row = q0 + i + half * 8;
      Out[((size_t)b * Sc + row) * Cc + (size_t)h * Dc + nt * 16 + l16] = o[nt][i];
    }
  }
}

extern "C" void kernel_launch(void* const* d_in, const int* in_sizes, int n_in,
                              void* d_out, int out_size, void* d_ws, size_t ws_size,
                              hipStream_t stream) {
  const float* Q   = (const float*)d_in[0];
  const float* K   = (const float*)d_in[1];
  const float* V   = (const float*)d_in[2];
  const float* sqk = (const float*)d_in[3];
  const float* zqk = (const float*)d_in[4];
  const float* sat = (const float*)d_in[5];
  const float* zat = (const float*)d_in[6];
  const float* sm  = (const float*)d_in[7];
  float* Out = (float*)d_out;

  dim3 grid(Bc * Hc * (Sc / 128));   // 512 workgroups
  dim3 block(256);                   // 8 wave32 waves
  mpattn_kernel<<<grid, block, 0, stream>>>(Q, K, V, sqk, zqk, sat, zat, sm, Out);

  (void)in_sizes; (void)n_in; (void)out_size; (void)d_ws; (void)ws_size;
}
